// Local_EncoderLayer_60215441490188
// MI455X (gfx1250) — matrix operand order repaired and checked by reading, not compiled, not run on hardware
//
#include <hip/hip_runtime.h>

typedef unsigned int   u32;
typedef unsigned short u16;

typedef __attribute__((ext_vector_type(16))) __bf16 v16bf;
typedef __attribute__((ext_vector_type(8)))  float  v8f;

union Frag { v16bf v; u32 u[8]; };

__device__ __forceinline__ u16 f2bf(float f) {
  union { float f; u32 u; } x; x.f = f;
  u32 r = x.u + 0x7FFFu + ((x.u >> 16) & 1u);   // round-to-nearest-even
  return (u16)(r >> 16);
}
__device__ __forceinline__ u32 f2bf2(float a, float b) {
  return (u32)f2bf(a) | ((u32)f2bf(b) << 16);
}
__device__ __forceinline__ float bf2f(u16 h) {
  union { u32 u; float f; } x; x.u = ((u32)h) << 16; return x.f;
}

// ---------------------------------------------------------------------------
// Tiled bf16 WMMA GEMM: A[M,K] bf16 row-major, Wt[N,K] bf16 (pre-transposed).
// Tile 128x128, BK=64 (2 WMMA k-steps per LDS stage), double-buffered.
// Epilogue: +bias, +resid(fp32), relu; writes fp32 C32 and/or bf16 C16.
// ---------------------------------------------------------------------------
__global__ __launch_bounds__(256)
void gemm_bf16(const u16* __restrict__ Ab, const u16* __restrict__ Wt,
               const float* __restrict__ bias, const float* __restrict__ resid,
               float* __restrict__ C32, u16* __restrict__ C16,
               int M, int N, int K, int relu)
{
  constexpr int BM = 128, BN = 128, BK = 64, LDP = BK + 8;   // 144B stride
  __shared__ u16 sA[2][BM][LDP];
  __shared__ u16 sB[2][BN][LDP];

  const int tid  = threadIdx.x;
  const int lane = tid & 31;
  const int wave = tid >> 5;
  const int wm   = wave & 3;                 // 32-row band
  const int wn   = wave >> 2;                // 64-col band
  const int half = lane >> 4;
  const int l16  = lane & 15;
  const long bM = (long)blockIdx.y * BM;
  const long bN = (long)blockIdx.x * BN;

  v8f acc[2][4];
#pragma unroll
  for (int i = 0; i < 2; ++i)
#pragma unroll
    for (int j = 0; j < 4; ++j)
#pragma unroll
      for (int v = 0; v < 8; ++v) acc[i][j][v] = 0.0f;

  const int nT = K / BK;

  auto stage = [&](int t, int buf) {
    const long k0 = (long)t * BK;
#pragma unroll
    for (int i = 0; i < 4; ++i) {
      const int vec = tid + 256 * i;                 // 0..1023
      const int r = vec >> 3, c = (vec & 7) << 3;    // 16B chunk
      *(uint4*)&sA[buf][r][c] = *(const uint4*)(Ab + (bM + r) * K + k0 + c);
    }
#pragma unroll
    for (int i = 0; i < 4; ++i) {
      const int vec = tid + 256 * i;
      const int r = vec >> 3, c = (vec & 7) << 3;
      *(uint4*)&sB[buf][r][c] = *(const uint4*)(Wt + (bN + r) * K + k0 + c);
    }
  };

  stage(0, 0);
  __syncthreads();
  for (int t = 0; t < nT; ++t) {
    const int cur = t & 1;
    if (t + 1 < nT) stage(t + 1, cur ^ 1);
    if (t + 2 < nT) {                                 // L2 prefetch hint
      const long kp = (long)(t + 2) * BK;
      const int r = tid >> 3, c = (tid & 7) << 3;
      __builtin_prefetch(Ab + (bM + r) * K + kp + c, 0, 1);
      __builtin_prefetch(Wt + (bN + r) * K + kp + c, 0, 1);
    }

#pragma unroll
    for (int ks = 0; ks < BK; ks += 32) {
      Frag af[2], bfr[4];
#pragma unroll
      for (int i = 0; i < 2; ++i) {
        const int row = wm * 32 + i * 16 + l16;
#pragma unroll
        for (int v = 0; v < 8; ++v) {
          const int kk = ks + 16 * half + 2 * v;
          af[i].u[v] = *(const u32*)&sA[cur][row][kk];
        }
      }
#pragma unroll
      for (int j = 0; j < 4; ++j) {
        const int col = wn * 64 + j * 16 + l16;
#pragma unroll
        for (int v = 0; v < 8; ++v) {
          const int kk = ks + 16 * half + 2 * v;
          bfr[j].u[v] = *(const u32*)&sB[cur][col][kk];
        }
      }
#pragma unroll
      for (int i = 0; i < 2; ++i)
#pragma unroll
        for (int j = 0; j < 4; ++j)
          acc[i][j] = __builtin_amdgcn_wmma_f32_16x16x32_bf16(
              false, af[i].v, false, bfr[j].v, (short)0, acc[i][j], false, false);
    }
    __syncthreads();
  }

#pragma unroll
  for (int i = 0; i < 2; ++i)
#pragma unroll
    for (int j = 0; j < 4; ++j) {
      const long col = bN + wn * 64 + j * 16 + l16;
      const float bv = bias ? bias[col] : 0.0f;
#pragma unroll
      for (int v = 0; v < 8; ++v) {
        const long row = bM + wm * 32 + i * 16 + 8 * half + v;
        float val = acc[i][j][v] + bv;
        if (resid) val += resid[row * N + col];
        if (relu)  val = fmaxf(val, 0.0f);
        if (C32) C32[row * N + col] = val;
        if (C16) C16[row * N + col] = f2bf(val);
      }
    }
}

// ---------------------------------------------------------------------------
// Attention for one (sequence, head): out = softmax(QK^T/8) @ V - V
// Q/K/V bf16 [16384, H*64]. AO bf16 head-major [row, h*64+v].
// ---------------------------------------------------------------------------
template <int S>
__global__ __launch_bounds__(256)
void attn(const u16* __restrict__ Q, const u16* __restrict__ Kt,
          const u16* __restrict__ V, u16* __restrict__ AO)
{
  __shared__ u16   sQ[S][72];
  __shared__ u16   sK[S][72];
  __shared__ u16   sVt[64][S + 8];     // sVt[v][s]
  __shared__ float sP[S][S + 1];
  __shared__ u16   sPb[S][S + 8];

  const int tid = threadIdx.x, lane = tid & 31, wave = tid >> 5;
  const int half = lane >> 4, l16 = lane & 15;
  const int seq = blockIdx.x >> 4;
  const int h   = blockIdx.x & 15;
  const long rowbase = (long)seq * S;
  const int  cbase   = h * 64;

  for (int idx = tid; idx < S * 8; idx += 256) {     // rows, 16B chunks
    const int r = idx >> 3, c = (idx & 7) << 3;
    const long g = (rowbase + r) * 1024 + cbase + c;
    *(uint4*)&sQ[r][c] = *(const uint4*)(Q + g);
    *(uint4*)&sK[r][c] = *(const uint4*)(Kt + g);
  }
  for (int idx = tid; idx < S * 64; idx += 256) {    // V transposed
    const int r = idx >> 6, c = idx & 63;
    sVt[c][r] = V[(rowbase + r) * 1024 + cbase + c];
  }
  __syncthreads();

  constexpr int MT = S / 16;
  // scores = Q K^T * (1/sqrt(64))
  for (int tile = wave; tile < MT * MT; tile += 8) {
    const int mt = (tile / MT) * 16, nt = (tile % MT) * 16;
    v8f acc;
#pragma unroll
    for (int v = 0; v < 8; ++v) acc[v] = 0.0f;
#pragma unroll
    for (int ks = 0; ks < 64; ks += 32) {
      Frag a, b;
#pragma unroll
      for (int v = 0; v < 8; ++v) {
        const int ka = ks + 16 * half + 2 * v;
        a.u[v] = *(const u32*)&sQ[mt + l16][ka];
        const int kb = ks + 16 * half + 2 * v;
        b.u[v] = *(const u32*)&sK[nt + l16][kb];
      }
      acc = __builtin_amdgcn_wmma_f32_16x16x32_bf16(false, a.v, false, b.v,
                                                    (short)0, acc, false, false);
    }
#pragma unroll
    for (int v = 0; v < 8; ++v)
      sP[mt + 8 * half + v][nt + l16] = acc[v] * 0.125f;
  }
  __syncthreads();

  // softmax rows
  for (int r = tid; r < S; r += 256) {
    float mx = -3.402823e38f;
    for (int c = 0; c < S; ++c) mx = fmaxf(mx, sP[r][c]);
    float sum = 0.0f;
    for (int c = 0; c < S; ++c) { const float e = __expf(sP[r][c] - mx); sP[r][c] = e; sum += e; }
    const float inv = 1.0f / sum;
    for (int c = 0; c < S; ++c) sPb[r][c] = f2bf(sP[r][c] * inv);
  }
  __syncthreads();

  // out = P @ V - V
  for (int tile = wave; tile < MT * 4; tile += 8) {
    const int mt = (tile >> 2) * 16, nt = (tile & 3) * 16;
    v8f acc;
#pragma unroll
    for (int v = 0; v < 8; ++v) acc[v] = 0.0f;
#pragma unroll
    for (int ks = 0; ks < S; ks += 32) {
      Frag a, b;
#pragma unroll
      for (int v = 0; v < 8; ++v) {
        const int ka = ks + 16 * half + 2 * v;
        a.u[v] = *(const u32*)&sPb[mt + l16][ka];
        const int kb = ks + 16 * half + 2 * v;
        b.u[v] = *(const u32*)&sVt[nt + l16][kb];
      }
      acc = __builtin_amdgcn_wmma_f32_16x16x32_bf16(false, a.v, false, b.v,
                                                    (short)0, acc, false, false);
    }
#pragma unroll
    for (int v = 0; v < 8; ++v) {
      const long row = rowbase + mt + 8 * half + v;
      const long g = row * 1024 + cbase + nt + l16;
      AO[g] = f2bf(acc[v] - bf2f(V[g]));
    }
  }
}

// ---------------------------------------------------------------------------
// Row LayerNorm over D=1024; writes fp32 (for residuals) + bf16 (for GEMM A)
// ---------------------------------------------------------------------------
__global__ __launch_bounds__(256)
void layernorm(const float* __restrict__ X, const float* __restrict__ g,
               const float* __restrict__ b, float* __restrict__ Yf,
               u16* __restrict__ Yb)
{
  __shared__ float red[256];
  const int tid = threadIdx.x;
  const long row = blockIdx.x;
  const float4 f = *(const float4*)(X + row * 1024 + tid * 4);
  red[tid] = f.x + f.y + f.z + f.w;
  __syncthreads();
  for (int o = 128; o > 0; o >>= 1) { if (tid < o) red[tid] += red[tid + o]; __syncthreads(); }
  const float mu = red[0] * (1.0f / 1024.0f);
  __syncthreads();
  const float dx = f.x - mu, dy = f.y - mu, dz = f.z - mu, dw = f.w - mu;
  red[tid] = dx * dx + dy * dy + dz * dz + dw * dw;
  __syncthreads();
  for (int o = 128; o > 0; o >>= 1) { if (tid < o) red[tid] += red[tid + o]; __syncthreads(); }
  const float inv = rsqrtf(red[0] * (1.0f / 1024.0f) + 1e-5f);
  const int c = tid * 4;
  float4 o4;
  o4.x = dx * inv * g[c + 0] + b[c + 0];
  o4.y = dy * inv * g[c + 1] + b[c + 1];
  o4.z = dz * inv * g[c + 2] + b[c + 2];
  o4.w = dw * inv * g[c + 3] + b[c + 3];
  *(float4*)(Yf + row * 1024 + c) = o4;
  uint2 p; p.x = f2bf2(o4.x, o4.y); p.y = f2bf2(o4.z, o4.w);
  *(uint2*)(Yb + row * 1024 + c) = p;
}

// ---------------------------------------------------------------------------
// Copy / permute kernels (fp32 + bf16 shadow)
// ---------------------------------------------------------------------------
__global__ __launch_bounds__(256)
void conv_copy(const float* __restrict__ X, float* __restrict__ Yf,
               u16* __restrict__ Yb)
{
  const long idx = ((long)blockIdx.x * 256 + threadIdx.x) * 4;
  const float4 f = *(const float4*)(X + idx);
  *(float4*)(Yf + idx) = f;
  uint2 p; p.x = f2bf2(f.x, f.y); p.y = f2bf2(f.z, f.w);
  *(uint2*)(Yb + idx) = p;
}

__global__ __launch_bounds__(256)
void regroup_conv(const float* __restrict__ X, float* __restrict__ Yf,
                  u16* __restrict__ Yb)
{
  const long r2 = blockIdx.x;                 // dst row: b*2048 + j*64 + i
  const int  bb = (int)(r2 >> 11);
  const int  rem = (int)(r2 & 2047);
  const int  j = rem >> 6, i = rem & 63;
  const long r1 = ((long)bb << 11) + i * 32 + j;
  const float4 f = ((const float4*)(X + r1 * 1024))[threadIdx.x];
  ((float4*)(Yf + r2 * 1024))[threadIdx.x] = f;
  uint2 p; p.x = f2bf2(f.x, f.y); p.y = f2bf2(f.z, f.w);
  ((uint2*)(Yb + r2 * 1024))[threadIdx.x] = p;
}

__global__ __launch_bounds__(256)
void ungroup(const float* __restrict__ X, float* __restrict__ Y)
{
  const long t = blockIdx.x;                  // out row: b*2048 + i*32 + j
  const int  bb = (int)(t >> 11);
  const int  rem = (int)(t & 2047);
  const int  i = rem >> 5, j = rem & 31;
  const long r = ((long)bb << 11) + j * 64 + i;
  ((float4*)(Y + t * 1024))[threadIdx.x] = ((const float4*)(X + r * 1024))[threadIdx.x];
}

// ---------------------------------------------------------------------------
// Weight packing: fp32 -> bf16, stored TRANSPOSED Wt[N][K]
// ---------------------------------------------------------------------------
__global__ __launch_bounds__(256)
void pack_qkv_t(const float* __restrict__ W, u16* __restrict__ Wt)
{
  // src [H=16, D=1024, 64]; dst Wt[n=h*64+kk][d], id = n*1024 + d
  const int id = blockIdx.x * 256 + threadIdx.x;
  const int n = id >> 10, d = id & 1023, h = n >> 6, kk = n & 63;
  Wt[id] = f2bf(W[((long)h * 1024 + d) * 64 + kk]);
}

__global__ __launch_bounds__(256)
void pack_t(const float* __restrict__ W, u16* __restrict__ Wt, int K, int N)
{
  // src W[K][N]; dst Wt[n][k], id = n*K + k
  const long id = (long)blockIdx.x * 256 + threadIdx.x;
  const int n = (int)(id / K), k = (int)(id % K);
  Wt[id] = f2bf(W[(long)k * N + n]);
}

// ---------------------------------------------------------------------------
// Host side
// ---------------------------------------------------------------------------
static void run_stage(hipStream_t s, float* xf, u16* xb, float* tmpf,
                      u16* Qb, u16* Kb, u16* Vb, u16* AO, u16* Hb,
                      const u16* wq, const u16* wk, const u16* wv, const u16* wp,
                      const float* pb, const float* g1, const float* bln1,
                      const u16* w1, const float* b1, const u16* w2, const float* b2,
                      const float* g2, const float* bln2, int S)
{
  const int M = 16384, D = 1024, DI = 4096;
  dim3 blk(256);
  dim3 gD(D / 128, M / 128), gDI(DI / 128, M / 128);
  gemm_bf16<<<gD, blk, 0, s>>>(xb, wq, nullptr, nullptr, nullptr, Qb, M, D, D, 0);
  gemm_bf16<<<gD, blk, 0, s>>>(xb, wk, nullptr, nullptr, nullptr, Kb, M, D, D, 0);
  gemm_bf16<<<gD, blk, 0, s>>>(xb, wv, nullptr, nullptr, nullptr, Vb, M, D, D, 0);
  if (S == 32) attn<32><<<(M / 32) * 16, blk, 0, s>>>(Qb, Kb, Vb, AO);
  else         attn<64><<<(M / 64) * 16, blk, 0, s>>>(Qb, Kb, Vb, AO);
  gemm_bf16<<<gD, blk, 0, s>>>(AO, wp, pb, xf, tmpf, nullptr, M, D, D, 0);
  layernorm<<<M, blk, 0, s>>>(tmpf, g1, bln1, xf, xb);
  gemm_bf16<<<gDI, blk, 0, s>>>(xb, w1, b1, nullptr, nullptr, Hb, M, DI, D, 1);
  gemm_bf16<<<gD, blk, 0, s>>>(Hb, w2, b2, xf, tmpf, nullptr, M, D, DI, 0);
  layernorm<<<M, blk, 0, s>>>(tmpf, g2, bln2, xf, xb);
}

extern "C" void kernel_launch(void* const* d_in, const int* in_sizes, int n_in,
                              void* d_out, int out_size, void* d_ws, size_t ws_size,
                              hipStream_t stream)
{
  (void)in_sizes; (void)n_in; (void)out_size; (void)ws_size;
  const size_t MB = 1ull << 20;
  char* ws = (char*)d_ws;

  float* X0f = (float*)(ws);                 // 64 MB
  float* X1f = (float*)(ws + 64 * MB);       // 64 MB
  u16*   X0b = (u16*)(ws + 128 * MB);        // 32 MB
  u16*   X1b = (u16*)(ws + 160 * MB);        // 32 MB
  u16*   Qb  = (u16*)(ws + 192 * MB);        // 32 MB
  u16*   Kb  = (u16*)(ws + 224 * MB);        // 32 MB
  u16*   Vb  = (u16*)(ws + 256 * MB);        // 32 MB
  u16*   AO  = (u16*)(ws + 288 * MB);        // 32 MB
  u16*   Hb  = (u16*)(ws + 192 * MB);        // 128 MB, reuses Q/K/V/AO
  u16*   laQ = (u16*)(ws + 320 * MB);
  u16* laK = laQ + (1 << 20);
  u16* laV = laK + (1 << 20);
  u16* laP = laV + (1 << 20);
  u16* saQ = laP + (1 << 20);
  u16* saK = saQ + (1 << 20);
  u16* saV = saK + (1 << 20);
  u16* saP = saV + (1 << 20);
  u16* lf1 = saP + (1 << 20);
  u16* lf2 = lf1 + (4 << 20);
  u16* pf1 = lf2 + (4 << 20);
  u16* pf2 = pf1 + (4 << 20);

  // d_in order: 0 enc | 1-7 la_{wqs,wks,wvs,pw,pb,g,b} | 8-14 sa_* | 15-20 lf_* | 21-26 pf_*
  dim3 blk(256);
  pack_qkv_t<<<4096, blk, 0, stream>>>((const float*)d_in[1], laQ);
  pack_qkv_t<<<4096, blk, 0, stream>>>((const float*)d_in[2], laK);
  pack_qkv_t<<<4096, blk, 0, stream>>>((const float*)d_in[3], laV);
  pack_t<<<4096, blk, 0, stream>>>((const float*)d_in[4], laP, 1024, 1024);
  pack_qkv_t<<<4096, blk, 0, stream>>>((const float*)d_in[8], saQ);
  pack_qkv_t<<<4096, blk, 0, stream>>>((const float*)d_in[9], saK);
  pack_qkv_t<<<4096, blk, 0, stream>>>((const float*)d_in[10], saV);
  pack_t<<<4096, blk, 0, stream>>>((const float*)d_in[11], saP, 1024, 1024);
  pack_t<<<16384, blk, 0, stream>>>((const float*)d_in[15], lf1, 1024, 4096);
  pack_t<<<16384, blk, 0, stream>>>((const float*)d_in[17], lf2, 4096, 1024);
  pack_t<<<16384, blk, 0, stream>>>((const float*)d_in[21], pf1, 1024, 4096);
  pack_t<<<16384, blk, 0, stream>>>((const float*)d_in[23], pf2, 4096, 1024);

  conv_copy<<<16384, blk, 0, stream>>>((const float*)d_in[0], X0f, X0b);

  // Local stage (seq len 32)
  run_stage(stream, X0f, X0b, X1f, Qb, Kb, Vb, AO, Hb,
            laQ, laK, laV, laP,
            (const float*)d_in[5], (const float*)d_in[6], (const float*)d_in[7],
            lf1, (const float*)d_in[16], lf2, (const float*)d_in[18],
            (const float*)d_in[19], (const float*)d_in[20], 32);

  regroup_conv<<<16384, blk, 0, stream>>>(X0f, X1f, X1b);

  // Global stage (seq len 64)
  run_stage(stream, X1f, X1b, X0f, Qb, Kb, Vb, AO, Hb,
            saQ, saK, saV, saP,
            (const float*)d_in[12], (const float*)d_in[13], (const float*)d_in[14],
            pf1, (const float*)d_in[22], pf2, (const float*)d_in[24],
            (const float*)d_in[25], (const float*)d_in[26], 64);

  ungroup<<<16384, blk, 0, stream>>>(X1f, (float*)d_out);
}
